// GraphAttentionLayer_22591527977462
// MI455X (gfx1250) — matrix operand order repaired and checked by reading, not compiled, not run on hardware
//
#include <hip/hip_runtime.h>
#include <hip/hip_bf16.h>

#define N_NODES 8192
#define F_IN    256
#define F_OUT   128

typedef __attribute__((ext_vector_type(2)))  float  v2f;
typedef __attribute__((ext_vector_type(8)))  float  v8f;
typedef __attribute__((ext_vector_type(16))) __bf16 v16bf;

// ---------------------------------------------------------------------------
// Kernel 1: h = X @ W + b   via V_WMMA_F32_16X16X4_F32.
// Grid: one block per 16-row strip; 8 waves/block, one 16-col tile per wave.
// Writes h in f32 row-major (for score dots) and bf16 transposed (f-major,
// 2 MB -> L2-resident B matrix for the fused attention GEMM).
// A layout (f32 16x4): lanes0-15 V0=K0,V1=K1 ; lanes16-31 V0=K2,V1=K3.
// B layout (f32 4x16): mirrored (lanes0-15 K0/K1, lanes16-31 K2/K3).
// C layout: lane L col N=L&15, VGPR v row M = v + 8*(L>=16).
// ---------------------------------------------------------------------------
__global__ __launch_bounds__(256)
void gat_h_gemm(const float* __restrict__ X, const float* __restrict__ W,
                const float* __restrict__ Wb, float* __restrict__ hF,
                __bf16* __restrict__ hT)
{
    const int lane = threadIdx.x & 31;
    const int wav  = threadIdx.x >> 5;          // 0..7 -> column tile
    const int m    = lane & 15;
    const int hi   = lane >> 4;
    const int r0   = blockIdx.x * 16;
    const int col  = wav * 16 + m;

    v8f c = {0.f, 0.f, 0.f, 0.f, 0.f, 0.f, 0.f, 0.f};

    const float* xrow = X + (size_t)(r0 + m) * F_IN;

    #pragma unroll 4
    for (int k0 = 0; k0 < F_IN; k0 += 4) {
        v2f a = *(const v2f*)(xrow + k0 + 2 * hi);               // K pair
        v2f b = { W[(size_t)(k0 + 2 * hi)     * F_OUT + col],
                  W[(size_t)(k0 + 2 * hi + 1) * F_OUT + col] };
        c = __builtin_amdgcn_wmma_f32_16x16x4_f32(
                false, a, false, b, (short)0, c, false, false);
    }

    const float bias = Wb[col];
    #pragma unroll
    for (int v = 0; v < 8; ++v) {
        const int row = r0 + v + 8 * hi;
        const float val = c[v] + bias;
        hF[(size_t)row * F_OUT + col]   = val;
        hT[(size_t)col * N_NODES + row] = (__bf16)val;
    }
}

// ---------------------------------------------------------------------------
// Kernel 2: s_src = h@a1, s_dst = h@a2  (trivial, one node per thread)
// ---------------------------------------------------------------------------
__global__ __launch_bounds__(256)
void gat_scores(const float* __restrict__ hF, const float* __restrict__ aw,
                float* __restrict__ ssrc, float* __restrict__ sdst)
{
    const int i = blockIdx.x * blockDim.x + threadIdx.x;
    const float* hrow = hF + (size_t)i * F_OUT;
    float s1 = 0.f, s2 = 0.f;
    #pragma unroll 8
    for (int f = 0; f < F_OUT; ++f) {
        const float hv = hrow[f];
        s1 = fmaf(hv, aw[f],         s1);
        s2 = fmaf(hv, aw[F_OUT + f], s2);
    }
    ssrc[i] = s1;
    sdst[i] = s2;
}

// ---------------------------------------------------------------------------
// Kernel 3: fused masked-softmax + attention GEMM (single pass over adj).
// w_ij = adj>0 ? exp(relu(ssrc_i + sdst_j + ab)) : 0   (softmax shift-invariant,
// relu output bounded -> no max-subtraction / rescaling needed)
// h'_i = (sum_j w_ij * h_j) / (sum_j w_ij)
// One wave owns 16 rows; j loop in chunks of 32 -> one bf16 A-fragment +
// 8x V_WMMA_F32_16X16X32_BF16 (covering all 128 features) per chunk.
// A layout (bf16 16x32): lane m (0-15): elems0-7 = K hi*16+0..7,
//                        elems8-15 = K hi*16+8+0..7  (hi = lane>>4).
// B layout (bf16 32x16): lane col N=L&15; elem e = K hi*16+e -> 32 contiguous
//                        bytes of f-major hT per lane.
// ---------------------------------------------------------------------------
__global__ __launch_bounds__(64)
void gat_attn(const int* __restrict__ adj, const __bf16* __restrict__ hT,
              const float* __restrict__ ssrc, const float* __restrict__ sdst,
              const float* __restrict__ abp, float* __restrict__ out)
{
    const int lane = threadIdx.x & 31;
    const int wav  = threadIdx.x >> 5;          // 2 waves/block
    const int m    = lane & 15;
    const int hi   = lane >> 4;
    const int r0   = (blockIdx.x * 2 + wav) * 16;
    const int row  = r0 + m;

    const float sm = ssrc[row];
    const float ab = abp[0];
    const int* __restrict__ arow = adj + (size_t)row * N_NODES;

    v8f acc[8];
    #pragma unroll
    for (int t = 0; t < 8; ++t)
        acc[t] = (v8f){0.f, 0.f, 0.f, 0.f, 0.f, 0.f, 0.f, 0.f};

    float wsum = 0.f;

    for (int j0 = 0; j0 < N_NODES; j0 += 32) {
        if (j0 + 512 < N_NODES)
            __builtin_prefetch(arow + j0 + 512, 0, 1);   // global_prefetch_b8

        const int kA = j0 + hi * 16;                      // this lane's low-K run
        const int kB = j0 + hi * 16 + 8;                  // this lane's high-K run
        const int4   p0 = *(const int4*)  (arow + kA);
        const int4   p1 = *(const int4*)  (arow + kA + 4);
        const int4   p2 = *(const int4*)  (arow + kB);
        const int4   p3 = *(const int4*)  (arow + kB + 4);
        const float4 q0 = *(const float4*)(sdst + kA);
        const float4 q1 = *(const float4*)(sdst + kA + 4);
        const float4 q2 = *(const float4*)(sdst + kB);
        const float4 q3 = *(const float4*)(sdst + kB + 4);

        v16bf af;
        #define WCALC(slot, av, sv)                         \
            {                                               \
                float z = sm + (sv) + ab;                   \
                z = z > 0.f ? z : 0.f;                      \
                float w = ((av) > 0) ? __expf(z) : 0.f;     \
                wsum += w;                                  \
                af[slot] = (__bf16)w;                       \
            }
        WCALC( 0, p0.x, q0.x) WCALC( 1, p0.y, q0.y)
        WCALC( 2, p0.z, q0.z) WCALC( 3, p0.w, q0.w)
        WCALC( 4, p1.x, q1.x) WCALC( 5, p1.y, q1.y)
        WCALC( 6, p1.z, q1.z) WCALC( 7, p1.w, q1.w)
        WCALC( 8, p2.x, q2.x) WCALC( 9, p2.y, q2.y)
        WCALC(10, p2.z, q2.z) WCALC(11, p2.w, q2.w)
        WCALC(12, p3.x, q3.x) WCALC(13, p3.y, q3.y)
        WCALC(14, p3.z, q3.z) WCALC(15, p3.w, q3.w)
        #undef WCALC

        #pragma unroll
        for (int t = 0; t < 8; ++t) {
            const v16bf bf = *(const v16bf*)(hT + (size_t)(t * 16 + m) * N_NODES
                                                + j0 + hi * 16);
            acc[t] = __builtin_amdgcn_wmma_f32_16x16x32_bf16(
                         false, af, false, bf, (short)0, acc[t], false, false);
        }
    }

    // rowsum: lanes L and L^16 hold disjoint K halves of the same row
    const float tot = wsum + __shfl_xor(wsum, 16, 32);
    const float inv = tot > 0.f ? 1.0f / tot : 0.f;      // inv(row m) in lanes m, m+16

    #pragma unroll
    for (int v = 0; v < 8; ++v) {
        const float invv = __shfl(inv, v + 8 * hi, 32);  // inv for row (v + 8*hi)
        const int orow = r0 + v + 8 * hi;
        float* orow_p = out + (size_t)orow * F_OUT;
        #pragma unroll
        for (int t = 0; t < 8; ++t)
            orow_p[t * 16 + m] = acc[t][v] * invv;
    }
}

// ---------------------------------------------------------------------------
extern "C" void kernel_launch(void* const* d_in, const int* in_sizes, int n_in,
                              void* d_out, int out_size, void* d_ws, size_t ws_size,
                              hipStream_t stream)
{
    const float* X   = (const float*)d_in[0];   // (8192, 256)
    const int*   adj = (const int*)  d_in[1];   // (8192, 8192)
    const float* Ww  = (const float*)d_in[2];   // (256, 128)
    const float* Wb  = (const float*)d_in[3];   // (128,)
    const float* aw  = (const float*)d_in[4];   // (256,)
    const float* ab  = (const float*)d_in[5];   // scalar
    float* out = (float*)d_out;                 // (8192, 128) f32

    char* ws = (char*)d_ws;                     // ~6.1 MB used
    float*  hF   = (float*) (ws);                                   // 4 MB
    __bf16* hT   = (__bf16*)(ws + 4u * 1024u * 1024u);              // 2 MB
    float*  ssrc = (float*) (ws + 6u * 1024u * 1024u);              // 32 KB
    float*  sdst = (float*) (ws + 6u * 1024u * 1024u + 32u * 1024u);// 32 KB

    gat_h_gemm<<<N_NODES / 16, 256, 0, stream>>>(X, Ww, Wb, hF, hT);
    gat_scores<<<N_NODES / 256, 256, 0, stream>>>(hF, aw, ssrc, sdst);
    gat_attn  <<<N_NODES / 32, 64, 0, stream>>>(adj, hT, ssrc, sdst, ab, out);
}
